// TemporalDownsamplingModel_63677185130799
// MI455X (gfx1250) — compile-verified, matrix operand order checked by reading, not run on hardware
//
#include <hip/hip_runtime.h>
#include <stdint.h>

typedef __attribute__((ext_vector_type(16))) __bf16 v16bf;
typedef __attribute__((ext_vector_type(8)))  __bf16 v8bf;
typedef __attribute__((ext_vector_type(8)))  float  v8f;

#define BB 32
#define SS 4096
#define FF 20
#define TT 512
#define HPAD 264
#define KRES 3   // k-steps whose Whh fragments stay register-resident

__device__ __forceinline__ float sigf(float x) { return 1.0f / (1.0f + __expf(-x)); }

// ---------------------------------------------------------------------------
// f32 -> bf16 conversion (weights)
// ---------------------------------------------------------------------------
__global__ void f32_to_bf16_k(const float* __restrict__ in, __bf16* __restrict__ out, int n) {
  int i = blockIdx.x * blockDim.x + threadIdx.x;
  if (i < n) out[i] = (__bf16)in[i];
}

// ---------------------------------------------------------------------------
// Importance scorer convs (f32 VALU; bandwidth-trivial)
// ---------------------------------------------------------------------------
__global__ void imp_conv1_k(const float* __restrict__ x, const float* __restrict__ w,
                            const float* __restrict__ bias, float* __restrict__ h1) {
  int id = blockIdx.x * blockDim.x + threadIdx.x;            // B*S*128
  if (id >= BB * SS * 128) return;
  int c = id & 127, s = (id >> 7) & (SS - 1), b = id >> 19;
  float acc = bias[c];
  for (int dk = 0; dk < 3; dk++) {
    int sp = s + dk - 1;
    if (sp < 0 || sp >= SS) continue;
    const float* xr = x + ((size_t)b * SS + sp) * FF;
    const float* wr = w + (c * FF) * 3 + dk;
    for (int f = 0; f < FF; f++) acc += xr[f] * wr[f * 3];
  }
  h1[(size_t)id] = fmaxf(acc, 0.0f);
}

__global__ void imp_conv2_k(const float* __restrict__ h1, const float* __restrict__ w,
                            const float* __restrict__ bias, float* __restrict__ h2) {
  int id = blockIdx.x * blockDim.x + threadIdx.x;            // B*S*64
  if (id >= BB * SS * 64) return;
  int c = id & 63, s = (id >> 6) & (SS - 1), b = id >> 18;
  float acc = bias[c];
  for (int dk = 0; dk < 3; dk++) {
    int sp = s + dk - 1;
    if (sp < 0 || sp >= SS) continue;
    const float* hr = h1 + ((size_t)b * SS + sp) * 128;
    const float* wr = w + (c * 128) * 3 + dk;
    for (int cc = 0; cc < 128; cc++) acc += hr[cc] * wr[cc * 3];
  }
  h2[(size_t)id] = fmaxf(acc, 0.0f);
}

__global__ void imp_conv3_k(const float* __restrict__ h2, const float* __restrict__ w,
                            const float* __restrict__ bias, float* __restrict__ imp) {
  int id = blockIdx.x * blockDim.x + threadIdx.x;            // B*S
  if (id >= BB * SS) return;
  const float* hr = h2 + (size_t)id * 64;
  float acc = bias[0];
  for (int c = 0; c < 64; c++) acc += hr[c] * w[c];
  imp[id] = sigf(acc);
}

// ---------------------------------------------------------------------------
// Multi-scale convs (k = 3,5,7,11), relu, output bf16 [B,S,256]
// ---------------------------------------------------------------------------
__global__ void msconv_k(const float* __restrict__ x,
                         const float* __restrict__ w3, const float* __restrict__ b3,
                         const float* __restrict__ w5, const float* __restrict__ b5,
                         const float* __restrict__ w7, const float* __restrict__ b7,
                         const float* __restrict__ w11, const float* __restrict__ b11,
                         __bf16* __restrict__ out) {
  int id = blockIdx.x * blockDim.x + threadIdx.x;            // B*S*256
  if (id >= BB * SS * 256) return;
  int c = id & 255, s = (id >> 8) & (SS - 1), b = id >> 20;
  int gi = c >> 6, ch = c & 63;
  const float *w, *bb; int K;
  if (gi == 0)      { w = w3;  bb = b3;  K = 3; }
  else if (gi == 1) { w = w5;  bb = b5;  K = 5; }
  else if (gi == 2) { w = w7;  bb = b7;  K = 7; }
  else              { w = w11; bb = b11; K = 11; }
  float acc = bb[ch];
  int hK = K >> 1;
  for (int dk = 0; dk < K; dk++) {
    int sp = s + dk - hK;
    if (sp < 0 || sp >= SS) continue;
    const float* xr = x + ((size_t)b * SS + sp) * FF;
    const float* wr = w + (ch * FF) * K + dk;
    for (int f = 0; f < FF; f++) acc += xr[f] * wr[f * K];
  }
  out[(size_t)id] = (__bf16)fmaxf(acc, 0.0f);
}

// ---------------------------------------------------------------------------
// Exact top-512 of 4096 per batch via bitonic sort in LDS, then sort indices
// ---------------------------------------------------------------------------
__global__ __launch_bounds__(1024) void topk_sort_k(const float* __restrict__ imp,
                                                    int* __restrict__ idx, float* __restrict__ xp) {
  __shared__ float sv[SS];
  __shared__ int   si[SS];
  int b = blockIdx.x, tid = threadIdx.x;
  const float* ib = imp + (size_t)b * SS;
  for (int p = 0; p < 4; p++) { int i = tid + p * 1024; sv[i] = ib[i]; si[i] = i; }
  __syncthreads();
  for (int k = 2; k <= SS; k <<= 1)
    for (int j = k >> 1; j > 0; j >>= 1) {
      for (int p = 0; p < 4; p++) {
        int i = tid + p * 1024;
        int ixj = i ^ j;
        if (ixj > i) {
          float v1 = sv[i], v2 = sv[ixj];
          int a1 = si[i], a2 = si[ixj];
          bool first = (v1 > v2) || (v1 == v2 && a1 < a2);   // descending rank
          bool desc = ((i & k) == 0);
          if (desc ? !first : first) { sv[i] = v2; sv[ixj] = v1; si[i] = a2; si[ixj] = a1; }
        }
      }
      __syncthreads();
    }
  // si[0..511] = top-512 indices; sort ascending
  for (int k = 2; k <= TT; k <<= 1)
    for (int j = k >> 1; j > 0; j >>= 1) {
      if (tid < TT) {
        int i = tid, ixj = i ^ j;
        if (ixj > i) {
          int a1 = si[i], a2 = si[ixj];
          bool asc = ((i & k) == 0);
          if (asc ? (a1 > a2) : (a1 < a2)) { si[i] = a2; si[ixj] = a1; }
        }
      }
      __syncthreads();
    }
  if (tid < TT) { idx[b * TT + tid] = si[tid]; xp[b * TT + tid] = (float)si[tid]; }
}

__global__ void gather_rows_k(const __bf16* __restrict__ comb, const int* __restrict__ idx,
                              __bf16* __restrict__ x0) {
  int id = blockIdx.x * blockDim.x + threadIdx.x;            // B*512*256
  if (id >= BB * TT * 256) return;
  int c = id & 255, j = (id >> 8) & (TT - 1), b = id >> 17;
  int s = idx[b * TT + j];
  x0[(size_t)id] = comb[((size_t)b * SS + s) * 256 + c];
}

// ---------------------------------------------------------------------------
// bf16 WMMA fragment loads (A row-major [M,K]; W row-major [N,K] == B^T)
// 16-bit A 16x32 layout per ISA: lanes 0-15 K{0..7,16..23}, lanes 16-31 K{8..15,24..31}
// ---------------------------------------------------------------------------
__device__ __forceinline__ v16bf load_frag16(const __bf16* base, int ld, int row, int k0,
                                             int half, bool valid) {
  v16bf v;
  if (valid) {
    const __bf16* p = base + (size_t)row * ld + k0 + half * 8;
    v8bf lo = *(const v8bf*)p;
    v8bf hi = *(const v8bf*)(p + 16);
#pragma unroll
    for (int e = 0; e < 8; e++) { v[e] = lo[e]; v[e + 8] = hi[e]; }
  } else {
#pragma unroll
    for (int e = 0; e < 16; e++) v[e] = (__bf16)0.0f;
  }
  return v;
}

// ---------------------------------------------------------------------------
// Generic GEMM: C[M,N] = A[M,K](bf16) * W[N,K]^T(bf16) + bias1 + bias2
// One wave per 16x16 tile; M multiple of 16; K multiple of 32; N arbitrary.
// ---------------------------------------------------------------------------
template <int RELU, int OUT_BF16>
__global__ __launch_bounds__(256) void gemm_bf16_wmma(
    const __bf16* __restrict__ A, const __bf16* __restrict__ W,
    const float* __restrict__ bias1, const float* __restrict__ bias2,
    float* __restrict__ Cf, __bf16* __restrict__ Cb, int M, int N, int K) {
  int lane = threadIdx.x & 31;
  int half = lane >> 4, r = lane & 15;
  int ntiles = (N + 15) >> 4;
  int mtiles = M >> 4;
  long tile = (long)blockIdx.x * 8 + (threadIdx.x >> 5);
  if (tile >= (long)mtiles * ntiles) return;
  int m0 = (int)(tile / ntiles) * 16;
  int n0 = (int)(tile % ntiles) * 16;
  v8f acc;
#pragma unroll
  for (int i = 0; i < 8; i++) acc[i] = 0.0f;
  bool bvalid = (n0 + r) < N;
  for (int k0 = 0; k0 < K; k0 += 32) {
    v16bf a = load_frag16(A, K, m0 + r, k0, half, true);
    v16bf b = load_frag16(W, K, n0 + r, k0, half, bvalid);
    acc = __builtin_amdgcn_wmma_f32_16x16x32_bf16(false, a, false, b, (short)0, acc, false, false);
  }
  int col = n0 + r;
  if (col < N) {
    float bs = (bias1 ? bias1[col] : 0.0f) + (bias2 ? bias2[col] : 0.0f);
#pragma unroll
    for (int i = 0; i < 8; i++) {
      int row = m0 + i + half * 8;
      float v = acc[i] + bs;
      if (RELU) v = fmaxf(v, 0.0f);
      if (OUT_BF16) Cb[(size_t)row * N + col] = (__bf16)v;
      else          Cf[(size_t)row * N + col] = v;
    }
  }
}

// ---------------------------------------------------------------------------
// Persistent bidirectional LSTM recurrent scan.
// Block = (batch tile of 16) x (direction). 16 waves (512 thr); wave w owns
// gate columns u in [w*16, w*16+16) for all 4 gates. Whh fragments for
// kk < KRES are pinned in registers (fits the per-wave VGPR budget at 4
// waves/SIMD); the rest are STREAMED from global (L2-resident, 512 KB)
// every step. The asm memory clobber stops the compiler from hoisting the
// streamed (loop-invariant) loads back into spilled registers.
// ---------------------------------------------------------------------------
__global__ __launch_bounds__(512) void lstm_scan_k(
    const __bf16* __restrict__ WhhF, const __bf16* __restrict__ WhhB,
    const float* __restrict__ gxF, const float* __restrict__ gxB,
    __bf16* __restrict__ out, int T) {
  __shared__ __bf16 hs[16 * HPAD];
  int dir = blockIdx.y;
  const __bf16* Whh = dir ? WhhB : WhhF;
  const float* gx = dir ? gxB : gxF;
  int b0 = blockIdx.x * 16;
  int tid = threadIdx.x, lane = tid & 31, wave = tid >> 5;
  int half = lane >> 4, r = lane & 15;
  int u = wave * 16 + r;                                  // this lane's unit col

  // Per-lane base into Whh; all tile/k offsets become constant immediates.
  const __bf16* bbase = Whh + (size_t)(wave * 16 + r) * 256 + half * 8;

  // Register-resident Whh fragments for kk = 0..KRES-1 (4 gates each).
  v16bf bfragR[KRES][4];
#pragma unroll
  for (int kk = 0; kk < KRES; kk++) {
#pragma unroll
    for (int gate = 0; gate < 4; gate++) {
      const __bf16* bp = bbase + (size_t)gate * 256 * 256 + kk * 32;
      v8bf lo = *(const v8bf*)bp;
      v8bf hi = *(const v8bf*)(bp + 16);
#pragma unroll
      for (int e = 0; e < 8; e++) { bfragR[kk][gate][e] = lo[e]; bfragR[kk][gate][e + 8] = hi[e]; }
    }
  }

  float creg[8];
#pragma unroll
  for (int i = 0; i < 8; i++) creg[i] = 0.0f;
  for (int i = tid; i < 16 * HPAD; i += 512) hs[i] = (__bf16)0.0f;
  __syncthreads();

#pragma unroll 1
  for (int step = 0; step < T; step++) {
    asm volatile("" ::: "memory");                // forbid hoisting streamed loads
    int t = dir ? (T - 1 - step) : step;
    v8f acc[4];
#pragma unroll
    for (int q = 0; q < 4; q++)
#pragma unroll
      for (int e = 0; e < 8; e++) acc[q][e] = 0.0f;

    // Phase 1: register-resident k-steps.
#pragma unroll
    for (int kk = 0; kk < KRES; kk++) {
      v16bf a;
      const __bf16* ap = hs + r * HPAD + kk * 32 + half * 8;
#pragma unroll
      for (int e = 0; e < 8; e++) { a[e] = ap[e]; a[e + 8] = ap[16 + e]; }
#pragma unroll
      for (int gate = 0; gate < 4; gate++) {
        acc[gate] = __builtin_amdgcn_wmma_f32_16x16x32_bf16(false, a, false, bfragR[kk][gate],
                                                            (short)0, acc[gate], false, false);
      }
    }
    // Phase 2: streamed k-steps (Whh from L2 each step; constant imm offsets).
#pragma unroll
    for (int kk = KRES; kk < 8; kk++) {
      v16bf a;
      const __bf16* ap = hs + r * HPAD + kk * 32 + half * 8;
#pragma unroll
      for (int e = 0; e < 8; e++) { a[e] = ap[e]; a[e + 8] = ap[16 + e]; }
#pragma unroll
      for (int gate = 0; gate < 4; gate++) {
        const __bf16* bp = bbase + (size_t)gate * 256 * 256 + kk * 32;
        v8bf lo = *(const v8bf*)bp;
        v8bf hi = *(const v8bf*)(bp + 16);
        v16bf bf;
#pragma unroll
        for (int e = 0; e < 8; e++) { bf[e] = lo[e]; bf[e + 8] = hi[e]; }
        acc[gate] = __builtin_amdgcn_wmma_f32_16x16x32_bf16(false, a, false, bf,
                                                            (short)0, acc[gate], false, false);
      }
    }
    __syncthreads();                              // all reads of hs done
#pragma unroll
    for (int i2 = 0; i2 < 8; i2++) {
      int row = i2 + half * 8;
      size_t gbase = ((size_t)(b0 + row) * T + t) * 1024;
      float ig = acc[0][i2] + gx[gbase + u];
      float fg = acc[1][i2] + gx[gbase + 256 + u];
      float gg = acc[2][i2] + gx[gbase + 512 + u];
      float og = acc[3][i2] + gx[gbase + 768 + u];
      float cc = sigf(fg) * creg[i2] + sigf(ig) * tanhf(gg);
      creg[i2] = cc;
      float hh = sigf(og) * tanhf(cc);
      __bf16 hv = (__bf16)hh;
      hs[row * HPAD + u] = hv;
      out[((size_t)(b0 + row) * T + t) * 512 + dir * 256 + u] = hv;
    }
    __syncthreads();                              // hs fully updated
  }
}

// ---------------------------------------------------------------------------
// LayerNorm(256) + affine + relu -> bf16
// ---------------------------------------------------------------------------
__global__ void layernorm_relu_k(const float* __restrict__ x, const float* __restrict__ g,
                                 const float* __restrict__ b, __bf16* __restrict__ out, int rows) {
  int row = blockIdx.x * blockDim.x + threadIdx.x;
  if (row >= rows) return;
  const float* xr = x + (size_t)row * 256;
  float m = 0.0f;
  for (int i = 0; i < 256; i++) m += xr[i];
  m *= (1.0f / 256.0f);
  float v = 0.0f;
  for (int i = 0; i < 256; i++) { float d = xr[i] - m; v += d * d; }
  v *= (1.0f / 256.0f);
  float inv = rsqrtf(v + 1e-5f);
  __bf16* o = out + (size_t)row * 256;
  for (int i = 0; i < 256; i++) {
    float t = (xr[i] - m) * inv * g[i] + b[i];
    o[i] = (__bf16)fmaxf(t, 0.0f);
  }
}

// ---------------------------------------------------------------------------
// Piecewise-linear interpolation back to [B,S,F] (== jnp.interp with clamping)
// ---------------------------------------------------------------------------
__global__ void interp_k(const float* __restrict__ xp, const float* __restrict__ dec,
                         float* __restrict__ out) {
  int id = blockIdx.x * blockDim.x + threadIdx.x;            // B*S
  if (id >= BB * SS) return;
  int s = id & (SS - 1), b = id >> 12;
  const float* xpb = xp + b * TT;
  float pos = (float)s;
  int j; float t;
  if (pos <= xpb[0]) { j = 0; t = 0.0f; }
  else if (pos >= xpb[TT - 1]) { j = TT - 2; t = 1.0f; }
  else {
    int lo = 0, hi = TT - 1;
    while (hi - lo > 1) { int mid = (lo + hi) >> 1; if (xpb[mid] <= pos) lo = mid; else hi = mid; }
    j = lo;
    float x0 = xpb[j], x1 = xpb[j + 1];
    t = (pos - x0) / (x1 - x0);
  }
  const float* d0 = dec + ((size_t)b * TT + j) * FF;
  const float* d1 = d0 + FF;
  float* o = out + (size_t)id * FF;
  for (int f = 0; f < FF; f++) o[f] = d0[f] + t * (d1[f] - d0[f]);
}

// ---------------------------------------------------------------------------
// Host orchestration
// ---------------------------------------------------------------------------
extern "C" void kernel_launch(void* const* d_in, const int* in_sizes, int n_in,
                              void* d_out, int out_size, void* d_ws, size_t ws_size,
                              hipStream_t stream) {
  (void)in_sizes; (void)n_in; (void)out_size; (void)ws_size;

  // Input mapping: setup_inputs() insertion order (sequence, then params dict).
  const float* seq    = (const float*)d_in[0];
  const float* imp_w1 = (const float*)d_in[1];
  const float* imp_b1 = (const float*)d_in[2];
  const float* imp_w2 = (const float*)d_in[3];
  const float* imp_b2 = (const float*)d_in[4];
  const float* imp_w3 = (const float*)d_in[5];
  const float* imp_b3 = (const float*)d_in[6];
  const float* ms_w[4]  = { (const float*)d_in[7],  (const float*)d_in[9],
                            (const float*)d_in[11], (const float*)d_in[13] };
  const float* ms_b[4]  = { (const float*)d_in[8],  (const float*)d_in[10],
                            (const float*)d_in[12], (const float*)d_in[14] };
  const float *Wih[3][2], *Whh[3][2], *bih[3][2], *bhh[3][2];
  for (int l = 0; l < 3; l++)
    for (int d = 0; d < 2; d++) {
      int base = 15 + (l * 2 + d) * 4;
      Wih[l][d] = (const float*)d_in[base + 0];
      Whh[l][d] = (const float*)d_in[base + 1];
      bih[l][d] = (const float*)d_in[base + 2];
      bhh[l][d] = (const float*)d_in[base + 3];
    }
  const float* dec_w1 = (const float*)d_in[39];
  const float* dec_b1 = (const float*)d_in[40];
  const float* ln_g   = (const float*)d_in[41];
  const float* ln_b   = (const float*)d_in[42];
  const float* dec_w2 = (const float*)d_in[43];
  const float* dec_b2 = (const float*)d_in[44];
  const float* dec_w3 = (const float*)d_in[45];
  const float* dec_b3 = (const float*)d_in[46];

  // Workspace carve-up
  uintptr_t cur = (uintptr_t)d_ws;
  auto alloc = [&](size_t bytes) -> void* {
    cur = (cur + 255) & ~(uintptr_t)255;
    void* p = (void*)cur;
    cur += bytes;
    return p;
  };
  const int M = BB * TT;                                    // 16384 rows
  float* h1   = (float*)alloc((size_t)BB * SS * 128 * 4);
  float* h2   = (float*)alloc((size_t)BB * SS * 64 * 4);
  float* impv = (float*)alloc((size_t)BB * SS * 4);
  __bf16* comb = (__bf16*)alloc((size_t)BB * SS * 256 * 2);
  int*   idx  = (int*)alloc((size_t)BB * TT * 4);
  float* xp   = (float*)alloc((size_t)BB * TT * 4);
  __bf16* x0  = (__bf16*)alloc((size_t)M * 256 * 2);
  __bf16* xa  = (__bf16*)alloc((size_t)M * 512 * 2);
  __bf16* xb  = (__bf16*)alloc((size_t)M * 512 * 2);
  float* gxf  = (float*)alloc((size_t)M * 1024 * 4);
  float* gxb  = (float*)alloc((size_t)M * 1024 * 4);
  __bf16* wihb[3][2]; __bf16* whhb[3][2];
  for (int l = 0; l < 3; l++) {
    int Din = (l == 0) ? 256 : 512;
    for (int d = 0; d < 2; d++) {
      wihb[l][d] = (__bf16*)alloc((size_t)1024 * Din * 2);
      whhb[l][d] = (__bf16*)alloc((size_t)1024 * 256 * 2);
    }
  }
  __bf16* dw1b = (__bf16*)alloc((size_t)256 * 512 * 2);
  __bf16* dw2b = (__bf16*)alloc((size_t)128 * 256 * 2);
  __bf16* dw3b = (__bf16*)alloc((size_t)20 * 128 * 2);
  float* y1   = (float*)alloc((size_t)M * 256 * 4);
  __bf16* lnq = (__bf16*)alloc((size_t)M * 256 * 2);
  __bf16* y2  = (__bf16*)alloc((size_t)M * 128 * 2);
  float* decb = (float*)alloc((size_t)M * FF * 4);

  auto cvt = [&](const float* src, __bf16* dst, int n) {
    f32_to_bf16_k<<<(n + 255) / 256, 256, 0, stream>>>(src, dst, n);
  };
  // Weight conversions
  for (int l = 0; l < 3; l++) {
    int Din = (l == 0) ? 256 : 512;
    for (int d = 0; d < 2; d++) {
      cvt(Wih[l][d], wihb[l][d], 1024 * Din);
      cvt(Whh[l][d], whhb[l][d], 1024 * 256);
    }
  }
  cvt(dec_w1, dw1b, 256 * 512);
  cvt(dec_w2, dw2b, 128 * 256);
  cvt(dec_w3, dw3b, 20 * 128);

  // Importance scorer
  { int n = BB * SS * 128; imp_conv1_k<<<(n + 255) / 256, 256, 0, stream>>>(seq, imp_w1, imp_b1, h1); }
  { int n = BB * SS * 64;  imp_conv2_k<<<(n + 255) / 256, 256, 0, stream>>>(h1, imp_w2, imp_b2, h2); }
  { int n = BB * SS;       imp_conv3_k<<<(n + 255) / 256, 256, 0, stream>>>(h2, imp_w3, imp_b3, impv); }

  // Multi-scale conv features
  { int n = BB * SS * 256;
    msconv_k<<<(n + 255) / 256, 256, 0, stream>>>(seq, ms_w[0], ms_b[0], ms_w[1], ms_b[1],
                                                  ms_w[2], ms_b[2], ms_w[3], ms_b[3], comb); }

  // Top-k + index sort, gather
  topk_sort_k<<<BB, 1024, 0, stream>>>(impv, idx, xp);
  { int n = BB * TT * 256; gather_rows_k<<<(n + 255) / 256, 256, 0, stream>>>(comb, idx, x0); }

  // 3-layer bidirectional LSTM
  const __bf16* xin = x0;
  int Din = 256;
  __bf16* bufs[3] = { xa, xb, xa };
  for (int l = 0; l < 3; l++) {
    for (int d = 0; d < 2; d++) {
      long tiles = (long)(M / 16) * (1024 / 16);
      int blocks = (int)((tiles + 7) / 8);
      gemm_bf16_wmma<0, 0><<<blocks, 256, 0, stream>>>(
          xin, wihb[l][d], bih[l][d], bhh[l][d], d ? gxb : gxf, nullptr, M, 1024, Din);
    }
    dim3 g(BB / 16, 2);
    lstm_scan_k<<<g, 512, 0, stream>>>(whhb[l][0], whhb[l][1], gxf, gxb, bufs[l], TT);
    xin = bufs[l];
    Din = 512;
  }

  // Decoder
  { long tiles = (long)(M / 16) * (256 / 16); int blocks = (int)((tiles + 7) / 8);
    gemm_bf16_wmma<0, 0><<<blocks, 256, 0, stream>>>(xin, dw1b, dec_b1, nullptr, y1, nullptr,
                                                     M, 256, 512); }
  layernorm_relu_k<<<(M + 255) / 256, 256, 0, stream>>>(y1, ln_g, ln_b, lnq, M);
  { long tiles = (long)(M / 16) * (128 / 16); int blocks = (int)((tiles + 7) / 8);
    gemm_bf16_wmma<1, 1><<<blocks, 256, 0, stream>>>(lnq, dw2b, dec_b2, nullptr, nullptr, y2,
                                                     M, 128, 256); }
  { long tiles = (long)(M / 16) * 2; int blocks = (int)((tiles + 7) / 8);
    gemm_bf16_wmma<0, 0><<<blocks, 256, 0, stream>>>(y2, dw3b, dec_b3, nullptr, decb, nullptr,
                                                     M, FF, 128); }

  // Interpolate back to [B,S,F]
  { int n = BB * SS; interp_k<<<(n + 255) / 256, 256, 0, stream>>>(xp, decb, (float*)d_out); }
}
